// CWL2GCNLayer_23184233464191
// MI455X (gfx1250) — matrix operand order repaired and checked by reading, not compiled, not run on hardware
//
#include <hip/hip_runtime.h>

typedef __attribute__((ext_vector_type(16))) _Float16 v16h;
typedef __attribute__((ext_vector_type(8)))  float    v8f;

#define DIM 64  // D_IN == D_OUT == 64

// ---------------------------------------------------------------------------
// WMMA fragment builders (wave32, 16x16x32 f16 -> f32)
// A (16x32): lane L(0..15) holds row M=L, K in [kb,kb+8) U [kb+16,kb+24),
//            kb = chunk*32 + (lane>=16 ? 8 : 0); element e<8 -> K=kb+e,
//            e>=8 -> K=kb+16+(e-8).
// B (32x16): lane holds col = nt*16 + (lane&15); same K mapping as A:
//            kb = chunk*32 + (lane>=16 ? 8 : 0); e<8 -> kb+e, e>=8 -> kb+16+(e-8).
// C/D (16x16 f32): VGPR i -> row = i + (lane>=16 ? 8 : 0), col = lane&15.
// ---------------------------------------------------------------------------
__device__ __forceinline__ v16h make_a_frag(const float* __restrict__ X,
                                            int row0, int chunk, int lane, int N) {
  int r = row0 + (lane & 15);
  if (r >= N) r = N - 1;                       // clamp (keeps EXEC all-1)
  int kb = chunk * 32 + ((lane & 16) ? 8 : 0);
  const float* p = X + (size_t)r * DIM + kb;
  v16h a;
#pragma unroll
  for (int i = 0; i < 8; ++i) a[i] = (_Float16)p[i];
#pragma unroll
  for (int i = 0; i < 8; ++i) a[8 + i] = (_Float16)p[16 + i];
  return a;
}

__device__ __forceinline__ v16h make_b_frag(const float* __restrict__ sW,
                                            int nt, int chunk, int lane) {
  int col = nt * 16 + (lane & 15);
  int kb  = chunk * 32 + ((lane & 16) ? 8 : 0);
  v16h b;
#pragma unroll
  for (int i = 0; i < 16; ++i) b[i] = (_Float16)sW[(kb + (i < 8 ? i : 8 + i)) * DIM + col];
  return b;
}

// ---------------------------------------------------------------------------
// Kernel 1: XW_prop = X @ W_prop   (one wave per 16-row tile, WMMA)
// ---------------------------------------------------------------------------
__global__ void __launch_bounds__(256) wl2_proj_prop(const float* __restrict__ X,
                                                     const float* __restrict__ Wp,
                                                     float* __restrict__ XWp, int N) {
  __shared__ float sW[DIM * DIM];
  for (int i = threadIdx.x; i < DIM * DIM; i += 256) sW[i] = Wp[i];
  __syncthreads();

  int wave = threadIdx.x >> 5;
  int lane = threadIdx.x & 31;
  int row0 = (blockIdx.x * 8 + wave) * 16;
  if (row0 >= N) return;                       // wave-uniform guard

  v16h a0 = make_a_frag(X, row0, 0, lane, N);
  v16h a1 = make_a_frag(X, row0, 1, lane, N);

  v8f acc[4];
#pragma unroll
  for (int nt = 0; nt < 4; ++nt) {
    v8f c = {};
    v16h b0 = make_b_frag(sW, nt, 0, lane);
    c = __builtin_amdgcn_wmma_f32_16x16x32_f16(false, a0, false, b0, (short)0, c, false, false);
    v16h b1 = make_b_frag(sW, nt, 1, lane);
    c = __builtin_amdgcn_wmma_f32_16x16x32_f16(false, a1, false, b1, (short)0, c, false, false);
    acc[nt] = c;
  }

  int mb = (lane & 16) ? 8 : 0;
  int cl = lane & 15;
#pragma unroll
  for (int nt = 0; nt < 4; ++nt)
#pragma unroll
    for (int i = 0; i < 8; ++i) {
      int r = row0 + mb + i;
      if (r < N) XWp[(size_t)r * DIM + nt * 16 + cl] = acc[nt][i];
    }
}

// ---------------------------------------------------------------------------
// Kernel 2: zero the conv accumulator (graph-capture safe)
// ---------------------------------------------------------------------------
__global__ void __launch_bounds__(256) wl2_zero(float4* __restrict__ p, long long n4) {
  long long t = (long long)blockIdx.x * 256 + threadIdx.x;
  if (t < n4) p[t] = float4{0.f, 0.f, 0.f, 0.f};
}

// ---------------------------------------------------------------------------
// Kernel 3 (dominant, L2-resident): gather two rows of XW_prop, fused
// add+bias+relu, atomic scatter-sum into conv. 16 threads per pair, float4.
// ---------------------------------------------------------------------------
__global__ void __launch_bounds__(256) wl2_pair_scatter(const float* __restrict__ XWp,
                                                        const int* __restrict__ ref_a,
                                                        const int* __restrict__ ref_b,
                                                        const int* __restrict__ backref,
                                                        const float* __restrict__ b_prop,
                                                        float* __restrict__ conv, int M) {
  long long t = (long long)blockIdx.x * 256 + threadIdx.x;
  if (t >= (long long)M * 16) return;
  int m = (int)(t >> 4);
  int c = ((int)t & 15) * 4;
  int ia = ref_a[m], ib = ref_b[m], ob = backref[m];

  float4 va = *(const float4*)(XWp + (size_t)ia * DIM + c);
  float4 vb = *(const float4*)(XWp + (size_t)ib * DIM + c);
  float4 bp = *(const float4*)(b_prop + c);
  float s0 = va.x + vb.x + bp.x; s0 = s0 > 0.f ? s0 : 0.f;
  float s1 = va.y + vb.y + bp.y; s1 = s1 > 0.f ? s1 : 0.f;
  float s2 = va.z + vb.z + bp.z; s2 = s2 > 0.f ? s2 : 0.f;
  float s3 = va.w + vb.w + bp.w; s3 = s3 > 0.f ? s3 : 0.f;

  float* dst = conv + (size_t)ob * DIM + c;
  atomicAdd(dst + 0, s0);
  atomicAdd(dst + 1, s1);
  atomicAdd(dst + 2, s2);
  atomicAdd(dst + 3, s3);
}

// ---------------------------------------------------------------------------
// Kernel 4: recompute XW and XW_back via WMMA, fuse with conv and output act:
// out = relu(XW + XW_back * conv + b)
// ---------------------------------------------------------------------------
__global__ void __launch_bounds__(256) wl2_output(const float* __restrict__ X,
                                                  const float* __restrict__ W,
                                                  const float* __restrict__ Wback,
                                                  const float* __restrict__ bias,
                                                  const float* __restrict__ conv,
                                                  float* __restrict__ out, int N) {
  __shared__ float sW[DIM * DIM];
  __shared__ float sWb[DIM * DIM];
  for (int i = threadIdx.x; i < DIM * DIM; i += 256) {
    sW[i]  = W[i];
    sWb[i] = Wback[i];
  }
  __syncthreads();

  int wave = threadIdx.x >> 5;
  int lane = threadIdx.x & 31;
  int row0 = (blockIdx.x * 8 + wave) * 16;
  if (row0 >= N) return;

  v16h a0 = make_a_frag(X, row0, 0, lane, N);
  v16h a1 = make_a_frag(X, row0, 1, lane, N);

  v8f accW[4], accB[4];
#pragma unroll
  for (int nt = 0; nt < 4; ++nt) {
    v8f cw = {};
    v16h b0 = make_b_frag(sW, nt, 0, lane);
    cw = __builtin_amdgcn_wmma_f32_16x16x32_f16(false, a0, false, b0, (short)0, cw, false, false);
    v16h b1 = make_b_frag(sW, nt, 1, lane);
    cw = __builtin_amdgcn_wmma_f32_16x16x32_f16(false, a1, false, b1, (short)0, cw, false, false);
    accW[nt] = cw;

    v8f cb = {};
    v16h d0 = make_b_frag(sWb, nt, 0, lane);
    cb = __builtin_amdgcn_wmma_f32_16x16x32_f16(false, a0, false, d0, (short)0, cb, false, false);
    v16h d1 = make_b_frag(sWb, nt, 1, lane);
    cb = __builtin_amdgcn_wmma_f32_16x16x32_f16(false, a1, false, d1, (short)0, cb, false, false);
    accB[nt] = cb;
  }

  int mb = (lane & 16) ? 8 : 0;
  int cl = lane & 15;
#pragma unroll
  for (int nt = 0; nt < 4; ++nt) {
    int col = nt * 16 + cl;
    float bb = bias[col];
#pragma unroll
    for (int i = 0; i < 8; ++i) {
      int r = row0 + mb + i;
      if (r < N) {
        float v = accW[nt][i] + accB[nt][i] * conv[(size_t)r * DIM + col] + bb;
        out[(size_t)r * DIM + col] = v > 0.f ? v : 0.f;
      }
    }
  }
}

// ---------------------------------------------------------------------------
// Host-side launcher
// Inputs: 0:X 1:ref_a 2:ref_b 3:backref 4:e_map 5:v_count 6:W 7:W_back
//         8:W_prop 9:b 10:b_prop
// ---------------------------------------------------------------------------
extern "C" void kernel_launch(void* const* d_in, const int* in_sizes, int n_in,
                              void* d_out, int out_size, void* d_ws, size_t ws_size,
                              hipStream_t stream) {
  const float* X       = (const float*)d_in[0];
  const int*   ref_a   = (const int*)d_in[1];
  const int*   ref_b   = (const int*)d_in[2];
  const int*   backref = (const int*)d_in[3];
  const float* W       = (const float*)d_in[6];
  const float* W_back  = (const float*)d_in[7];
  const float* W_prop  = (const float*)d_in[8];
  const float* b       = (const float*)d_in[9];
  const float* b_prop  = (const float*)d_in[10];

  const int N = in_sizes[0] / DIM;
  const int M = in_sizes[1];

  // Workspace: XW_prop [N*64] then conv [N*64]; both L2-resident (~102 MB).
  float* XWp  = (float*)d_ws;
  float* conv = XWp + (size_t)N * DIM;

  const int tiles   = (N + 15) / 16;
  const int gemmBlk = (tiles + 7) / 8;   // 8 waves (16-row tiles) per block

  wl2_proj_prop<<<gemmBlk, 256, 0, stream>>>(X, W_prop, XWp, N);

  long long n4 = ((long long)N * DIM) / 4;
  int zb = (int)((n4 + 255) / 256);
  wl2_zero<<<zb, 256, 0, stream>>>((float4*)conv, n4);

  long long pt = (long long)M * 16;
  int pb = (int)((pt + 255) / 256);
  wl2_pair_scatter<<<pb, 256, 0, stream>>>(XWp, ref_a, ref_b, backref, b_prop, conv, M);

  wl2_output<<<gemmBlk, 256, 0, stream>>>(X, W, W_back, b, conv, (float*)d_out, N);
}
